// ResRNN_Cell_68186900791473
// MI455X (gfx1250) — matrix operand order repaired and checked by reading, not compiled, not run on hardware
//
#include <hip/hip_runtime.h>
#include <hip/hip_bf16.h>

typedef __attribute__((ext_vector_type(16))) _Float16 v16h;
typedef __attribute__((ext_vector_type(8)))  _Float16 v8h;
typedef __attribute__((ext_vector_type(8)))  float    v8f;
typedef __attribute__((ext_vector_type(4)))  float    v4f;

#define BB 128
#define LL 512
#define DD 128
#define HH 1024
#define BH (BB * HH)   // 131072
#define NBLK_REC 64u

union V16U { v16h v; v8h h[2]; };

// ---------------------------------------------------------------------------
// Kernel 0: convert Wh (fp32) -> f16, zero h0 double buffers + barrier words.
// ---------------------------------------------------------------------------
__global__ __launch_bounds__(256) void k_init(const float* __restrict__ Wh,
                                              _Float16* __restrict__ Wh16,
                                              _Float16* __restrict__ h0a,
                                              _Float16* __restrict__ h0b,
                                              unsigned* __restrict__ bar) {
  int idx = blockIdx.x * 256 + threadIdx.x;
  if (idx < 2) bar[idx] = 0u;
  if (idx < HH * HH) Wh16[idx] = (_Float16)Wh[idx];
  if (idx < BB * HH) {
    h0a[idx] = (_Float16)0.f;
    h0b[idx] = (_Float16)0.f;
  }
}

// ---------------------------------------------------------------------------
// Kernel 1: input projection. out[l,b,h] = sum_d x[b,l,d]*Wi[h,d] + bi[h] + bh[h]
// grid = (H/16, L); block = 256 (8 waves = 8 b-tiles, B = 128 = 8*16).
// ---------------------------------------------------------------------------
__global__ __launch_bounds__(256) void k_inproj(const float* __restrict__ x,
                                                const float* __restrict__ Wi,
                                                const float* __restrict__ bi,
                                                const float* __restrict__ bh,
                                                float* __restrict__ out) {
  __shared__ _Float16 sB[16 * DD];   // Wi tile: [n_local][k], f16, 4 KB
  __shared__ float    sbias[16];

  const int n0 = blockIdx.x * 16;
  const int l  = blockIdx.y;
  const int tid = threadIdx.x;

  for (int t = tid; t < 16 * DD; t += 256) {
    int n = t >> 7, k = t & (DD - 1);
    sB[t] = (_Float16)Wi[(size_t)(n0 + n) * DD + k];
  }
  if (tid < 16) sbias[tid] = bi[n0 + tid] + bh[n0 + tid];
  __syncthreads();

  const int lane = tid & 31;
  const int b0   = (tid >> 5) * 16;
  const int m    = lane & 15;             // A row / C col index within tile
  const int hiHalf = (lane < 16) ? 0 : 1;
  const int koffa  = hiHalf ? 16 : 0;     // A: K sub-offset per lane group
  const int koffb  = hiHalf ? 16 : 0;     // B: K sub-offset per lane group
  const int mrow   = hiHalf ? 8 : 0;      // C/D: row offset per lane group

  v8f c;
  {
    float bias = sbias[m];
#pragma unroll
    for (int r = 0; r < 8; ++r) c[r] = bias;
  }

  const float* ap = x + ((size_t)(b0 + m) * LL + l) * DD;
#pragma unroll
  for (int kk = 0; kk < DD; kk += 32) {
    // A fragment: elems[0..15]=K kk+koffa..kk+koffa+15
    v4f a00 = *(const v4f*)(ap + kk + koffa);
    v4f a01 = *(const v4f*)(ap + kk + koffa + 4);
    v4f a10 = *(const v4f*)(ap + kk + koffa + 8);
    v4f a11 = *(const v4f*)(ap + kk + koffa + 12);
    v16h a;
#pragma unroll
    for (int j = 0; j < 4; ++j) {
      a[j]      = (_Float16)a00[j];
      a[4 + j]  = (_Float16)a01[j];
      a[8 + j]  = (_Float16)a10[j];
      a[12 + j] = (_Float16)a11[j];
    }
    v16h b = *(const v16h*)&sB[m * DD + kk + koffb];
    c = __builtin_amdgcn_wmma_f32_16x16x32_f16(false, a, false, b,
                                               (short)0, c, false, false);
  }

  float* op = out + (size_t)l * BH + (size_t)(b0 + mrow) * HH + n0 + m;
#pragma unroll
  for (int r = 0; r < 8; ++r) op[(size_t)r * HH] = c[r];
}

// ---------------------------------------------------------------------------
// Grid-wide barrier (sense via generation counter) on d_ws.
// ---------------------------------------------------------------------------
__device__ __forceinline__ void grid_barrier(unsigned* bar) {
  __syncthreads();
  if (threadIdx.x == 0) {
    __threadfence();
    volatile unsigned* vbar = (volatile unsigned*)bar;
    unsigned gen = vbar[1];
    if (atomicAdd(&bar[0], 1u) == NBLK_REC - 1u) {
      atomicExch(&bar[0], 0u);
      __threadfence();
      atomicAdd(&bar[1], 1u);
    } else {
      while (vbar[1] == gen) { __builtin_amdgcn_s_sleep(2); }
    }
  }
  __syncthreads();
}

// ---------------------------------------------------------------------------
// Kernel 2: persistent recurrent scan. 64 blocks x 256 threads = 512 waves,
// one 16x16 output tile of [B=128, H=1024] per wave. Wh N-slice in LDS.
// out[i] (already = xs+bh) is read as accumulator init; hn written back;
// tanh state update produces h0 (f16, double-buffered in global).
// ---------------------------------------------------------------------------
__global__ __launch_bounds__(256) void k_recur(float* __restrict__ out,
                                               const _Float16* __restrict__ Wh16,
                                               _Float16* __restrict__ h0a,
                                               _Float16* __restrict__ h0b,
                                               unsigned* __restrict__ bar) {
  __shared__ _Float16 sWh[16 * HH];   // 32 KB: rows n0..n0+15 of Wh, f16

  const int n0  = blockIdx.x * 16;
  const int tid = threadIdx.x;
  for (int t = tid; t < 16 * HH; t += 256) {
    sWh[t] = Wh16[(size_t)(n0 + (t >> 10)) * HH + (t & (HH - 1))];
  }
  __syncthreads();

  const int lane = tid & 31;
  const int b0   = (tid >> 5) * 16;
  const int m    = lane & 15;
  const int hiHalf = (lane < 16) ? 0 : 1;
  const int koffa  = hiHalf ? 16 : 0;
  const int koffb  = hiHalf ? 16 : 0;
  const int mrow   = hiHalf ? 8 : 0;

  float ht[8], hstart[8];
#pragma unroll
  for (int r = 0; r < 8; ++r) { ht[r] = 0.f; hstart[r] = 0.f; }

  _Float16* h0buf[2] = { h0a, h0b };
  const size_t crow = (size_t)(b0 + mrow) * HH + n0 + m;  // C/D base index
  const size_t arow = (size_t)(b0 + m) * HH;              // A base row in h0

#pragma unroll 1
  for (int i = 0; i < LL; ++i) {
    const _Float16* h0r = h0buf[i & 1];
    _Float16*       h0w = h0buf[(i + 1) & 1];
    float* op = out + (size_t)i * BH + crow;

    v8f c;
#pragma unroll
    for (int r = 0; r < 8; ++r) c[r] = op[(size_t)r * HH];  // xs + bh

    const _Float16* ap = h0r + arow;
#pragma unroll 4
    for (int kk = 0; kk < HH; kk += 32) {
      V16U ua;
      ua.h[0] = *(const v8h*)(ap + kk + koffa);
      ua.h[1] = *(const v8h*)(ap + kk + koffa + 8);
      v16h b = *(const v16h*)&sWh[m * HH + kk + koffb];
      c = __builtin_amdgcn_wmma_f32_16x16x32_f16(false, ua.v, false, b,
                                                 (short)0, c, false, false);
    }

    const bool is0   = (i == 0);
    const bool isL2  = (i == LL - 2);
    const bool ismod = ((i & 3) == 0) && !is0 && !isL2;
#pragma unroll
    for (int r = 0; r < 8; ++r) {
      float hn = c[r];
      op[(size_t)r * HH] = hn;                 // output is pre-tanh hn
      float nh0;
      if (is0)       { hstart[r] = hn; nh0 = 0.f; }   // h0 NOT updated at i==0
      else if (isL2) { nh0 = tanhf(hn + hstart[r]); }
      else if (ismod){ nh0 = tanhf(hn + ht[r]); ht[r] = hn; }
      else           { nh0 = tanhf(hn); }
      h0w[crow + (size_t)r * HH - 0] = (_Float16)nh0;
    }

    grid_barrier(bar);
  }
}

// ---------------------------------------------------------------------------
extern "C" void kernel_launch(void* const* d_in, const int* in_sizes, int n_in,
                              void* d_out, int out_size, void* d_ws, size_t ws_size,
                              hipStream_t stream) {
  (void)in_sizes; (void)n_in; (void)out_size; (void)ws_size;
  const float* x  = (const float*)d_in[0];
  const float* Wi = (const float*)d_in[1];
  const float* bi = (const float*)d_in[2];
  const float* Wh = (const float*)d_in[3];
  const float* bh = (const float*)d_in[4];
  float* out = (float*)d_out;

  char* ws = (char*)d_ws;
  unsigned* bar   = (unsigned*)ws;                                   // 256 B
  _Float16* Wh16  = (_Float16*)(ws + 256);                           // 2 MiB
  _Float16* h0a   = (_Float16*)(ws + 256 + (size_t)HH * HH * 2);     // 256 KiB
  _Float16* h0b   = (_Float16*)(ws + 256 + (size_t)HH * HH * 2
                                        + (size_t)BB * HH * 2);      // 256 KiB

  k_init<<<dim3((HH * HH + 255) / 256), dim3(256), 0, stream>>>(Wh, Wh16, h0a, h0b, bar);
  k_inproj<<<dim3(HH / 16, LL), dim3(256), 0, stream>>>(x, Wi, bi, bh, out);
  k_recur<<<dim3(NBLK_REC), dim3(256), 0, stream>>>(out, Wh16, h0a, h0b, bar);
}
